// RoPEMaskedAttentionHead_80384607911994
// MI455X (gfx1250) — compile-verified, matrix operand order checked by reading, not run on hardware
//
#include <hip/hip_runtime.h>
#include <hip/hip_bf16.h>

typedef __bf16 v16bf __attribute__((ext_vector_type(16)));
typedef __bf16 v8bf  __attribute__((ext_vector_type(8)));
typedef float  v8f   __attribute__((ext_vector_type(8)));
typedef int    v4i   __attribute__((ext_vector_type(4)));

#define B_  4
#define M_  2048
#define D_  1024

// ---- gfx1250 async global->LDS copy (guarded; falls back to ds copy)
#if defined(__has_builtin)
#if __has_builtin(__builtin_amdgcn_global_load_async_to_lds_b128)
#define HAVE_ASYNC_LDS 1
#endif
#endif

#if defined(HAVE_ASYNC_LDS)
#define ASYNC_LDS_CP(dst, src)                                              \
  __builtin_amdgcn_global_load_async_to_lds_b128(                           \
      (__attribute__((address_space(1))) v4i*)(void*)(src),                 \
      (__attribute__((address_space(3))) v4i*)(void*)(dst), 0, 0)
#if __has_builtin(__builtin_amdgcn_s_wait_asynccnt)
#define ASYNC_LDS_WAIT() __builtin_amdgcn_s_wait_asynccnt(0)
#else
#define ASYNC_LDS_WAIT() asm volatile("s_wait_asynccnt 0x0" ::: "memory")
#endif
#else
#define ASYNC_LDS_CP(dst, src) (*(v8bf*)(dst) = *(const v8bf*)(src))
#define ASYNC_LDS_WAIT() ((void)0)
#endif

__device__ __forceinline__ v8f wmma_bf16(v16bf a, v16bf b, v8f c) {
  return __builtin_amdgcn_wmma_f32_16x16x32_bf16(false, a, false, b,
                                                 (short)0, c, false, false);
}

__device__ __forceinline__ v8f zero_v8f() {
  v8f z;
#pragma unroll
  for (int i = 0; i < 8; ++i) z[i] = 0.f;
  return z;
}

// Assemble a 32B fragment from two 16B LDS loads (rows are only 16B aligned).
__device__ __forceinline__ v16bf lds_frag(const __bf16* p) {
  v8bf lo = *(const v8bf*)p;
  v8bf hi = *(const v8bf*)(p + 8);
  return __builtin_shufflevector(lo, hi, 0, 1, 2, 3, 4, 5, 6, 7,
                                 8, 9, 10, 11, 12, 13, 14, 15);
}

// ---------------------------------------------------------------------------
// Kernel 1: f32 -> bf16 conversion (vectorized, packed 8B stores)
// ---------------------------------------------------------------------------
__global__ void cvt_f32_bf16(const float* __restrict__ src,
                             unsigned short* __restrict__ dst_u, long n4) {
  long i = (long)blockIdx.x * blockDim.x + threadIdx.x;
  long stride = (long)gridDim.x * blockDim.x;
  for (; i < n4; i += stride) {
    float4 f = ((const float4*)src)[i];
    union { __bf16 h[4]; short4 s4; } u;
    u.h[0] = (__bf16)f.x; u.h[1] = (__bf16)f.y;
    u.h[2] = (__bf16)f.z; u.h[3] = (__bf16)f.w;
    ((short4*)dst_u)[i] = u.s4;
  }
}

// ---------------------------------------------------------------------------
// Kernel 2: Q/K/V = x @ w^T with weights async-staged in LDS, shared A frag,
// register-double-buffered A, fast-math RoPE epilogue.
// Writes Qr, Kr row-major and VT[b][d][m].
// Grid: (64 row-groups of 128, 64 col-tiles of 16); block = 256 (8 waves).
// ---------------------------------------------------------------------------
#define WPAD 1032  // 16 B-aligned rows, bank stride 4
__global__ __launch_bounds__(256)
void qkv_rope(const unsigned short* __restrict__ xb_u,
              const unsigned short* __restrict__ wqb_u,
              const unsigned short* __restrict__ wkb_u,
              const unsigned short* __restrict__ wvb_u,
              unsigned short* __restrict__ Qr_u,
              unsigned short* __restrict__ Kr_u,
              unsigned short* __restrict__ VT_u) {
  const __bf16* xb  = (const __bf16*)xb_u;
  const __bf16* wqb = (const __bf16*)wqb_u;
  const __bf16* wkb = (const __bf16*)wkb_u;
  const __bf16* wvb = (const __bf16*)wvb_u;
  __bf16* Qr = (__bf16*)Qr_u;
  __bf16* Kr = (__bf16*)Kr_u;
  __bf16* VT = (__bf16*)VT_u;

  __shared__ __align__(16) __bf16 WL[3][16][WPAD];   // ~99 KB of 320 KB/WGP

  const int tid = threadIdx.x;
  const int lane = tid & 31;
  const int wv_id = tid >> 5;
  const int ln = lane & 15;
  const int half = lane >> 4;

  const int rowBase = blockIdx.x * 128 + wv_id * 16;
  const int colBase = blockIdx.y * 16;
  const int col = colBase + ln;
  const int koff = half * 16;

  // ---- async-stage all three weight tiles (16 cols x 1024 k) into LDS
#pragma unroll
  for (int m = 0; m < 3; ++m) {
    const __bf16* wp = (m == 0) ? wqb : ((m == 1) ? wkb : wvb);
    for (int i = tid; i < 2048; i += 256) {      // 16 rows x 128 16B-chunks
      const int row = i >> 7, ch = i & 127;
      ASYNC_LDS_CP(&WL[m][row][ch * 8],
                   wp + (size_t)(colBase + row) * D_ + ch * 8);
    }
  }
  ASYNC_LDS_WAIT();
  __syncthreads();

  const __bf16* arow = xb + (size_t)(rowBase + ln) * D_ + koff;

  v8f cq = zero_v8f(), ck = zero_v8f(), cvv = zero_v8f();
  v16bf a_cur = *(const v16bf*)(arow);

#pragma unroll 4
  for (int k0 = 0; k0 < D_; k0 += 32) {
    v16bf a_nxt = a_cur;
    if (k0 + 32 < D_) a_nxt = *(const v16bf*)(arow + k0 + 32);  // prefetch A
    v16bf fq = lds_frag(&WL[0][ln][k0 + koff]);
    v16bf fk = lds_frag(&WL[1][ln][k0 + koff]);
    v16bf fv = lds_frag(&WL[2][ln][k0 + koff]);
    cq  = wmma_bf16(a_cur, fq, cq);
    ck  = wmma_bf16(a_cur, fk, ck);
    cvv = wmma_bf16(a_cur, fv, cvv);
    a_cur = a_nxt;
  }

  // RoPE: i = col/2, phi = 10000^(-2(i-1)/D), ang = pos*phi
  const float phi =
      __expf(-2.0f * ((float)(col >> 1) - 1.0f) * (9.210340371976184f / (float)D_));
  const bool evencol = ((col & 1) == 0);

#pragma unroll
  for (int r = 0; r < 8; ++r) {
    const int gr = rowBase + r + half * 8;
    const int pos = gr & (M_ - 1);
    const float ang = (float)pos * phi;
    const float c_ = __cosf(ang), s_ = __sinf(ang);   // v_cos_f32 / v_sin_f32

    float vq = cq[r];
    float pq = __shfl_xor(vq, 1, 32);
    float qo = evencol ? (vq * c_ + pq * s_) : (vq * c_ - pq * s_);
    Qr[(size_t)gr * D_ + col] = (__bf16)qo;

    float vk = ck[r];
    float pk = __shfl_xor(vk, 1, 32);
    float ko = evencol ? (vk * c_ + pk * s_) : (vk * c_ - pk * s_);
    Kr[(size_t)gr * D_ + col] = (__bf16)ko;

    const int bb = gr >> 11, mm = gr & (M_ - 1);
    VT[((size_t)bb * D_ + col) * (size_t)M_ + mm] = (__bf16)cvv[r];
  }
}

// ---------------------------------------------------------------------------
// Kernel 3: single-pass causal flash attention.
// Block = (batch, 16-query tile); 8 waves each own a 128-wide d-slice of O.
// Fragment loads issued up-front; next key tile prefetched; softmax stats
// parallelized across all 32 lanes of wave 0 with shfl reductions.
// ---------------------------------------------------------------------------
__global__ __launch_bounds__(256)
void attn_flash(const unsigned short* __restrict__ Qr_u,
                const unsigned short* __restrict__ Kr_u,
                const unsigned short* __restrict__ VT_u,
                float* __restrict__ out) {
  const __bf16* Qr = (const __bf16*)Qr_u;
  const __bf16* Kr = (const __bf16*)Kr_u;
  const __bf16* VT = (const __bf16*)VT_u;

  __shared__ float Sp[8][16][32];
  __shared__ float Sred[16][32];
  __shared__ __align__(32) __bf16 P[16][32];
  __shared__ float mrow[16], lrow[16], frow[16];

  const int tid = threadIdx.x;
  const int lane = tid & 31;
  const int w = tid >> 5;
  const int ln = lane & 15;
  const int half = lane >> 4;

  const int qt = blockIdx.x;
  const int b  = blockIdx.y;
  const int qbase = qt * 16;
  const size_t rowoff = (size_t)b * M_ + qbase;

  if (tid < 16) { mrow[tid] = -3.0e38f; lrow[tid] = 0.f; }
  __syncthreads();

  const int dbase = w * 128;
  const int koff = half * 16;

  v16bf aq[4];
  {
    const __bf16* qrow = Qr + (rowoff + ln) * D_ + dbase + koff;
#pragma unroll
    for (int s = 0; s < 4; ++s) aq[s] = *(const v16bf*)(qrow + s * 32);
  }

  v8f o[8];
#pragma unroll
  for (int j = 0; j < 8; ++j) o[j] = zero_v8f();

  const int nkt = (qbase + 16 + 31) >> 5;
  const float scale = 0.03125f;  // 1/sqrt(1024)

  for (int kt = 0; kt < nkt; ++kt) {
    const int kb = kt * 32;
    const __bf16* k0p = Kr + ((size_t)b * M_ + kb + ln) * D_ + dbase + koff;
    const __bf16* k1p = k0p + (size_t)16 * D_;
    const __bf16* vtp =
        VT + ((size_t)b * D_ + dbase + ln) * (size_t)M_ + kb + koff;

    // ---- issue all K-fragment loads, then prefetch next tile, then WMMA
    v16bf b0f[4], b1f[4];
#pragma unroll
    for (int s = 0; s < 4; ++s) {
      b0f[s] = *(const v16bf*)(k0p + s * 32);
      b1f[s] = *(const v16bf*)(k1p + s * 32);
    }
    if (kt + 1 < nkt) {
      const __bf16* nk0 = k0p + (size_t)32 * D_;
      const __bf16* nk1 = k1p + (size_t)32 * D_;
      __builtin_prefetch(nk0, 0, 1);
      __builtin_prefetch(nk0 + 64, 0, 1);
      __builtin_prefetch(nk1, 0, 1);
      __builtin_prefetch(nk1 + 64, 0, 1);
      __builtin_prefetch(vtp + 32, 0, 1);
      __builtin_prefetch(vtp + (size_t)4 * 16 * M_ + 32, 0, 1);
    }
    v8f s0 = zero_v8f(), s1 = zero_v8f();
#pragma unroll
    for (int s = 0; s < 4; ++s) {
      s0 = wmma_bf16(aq[s], b0f[s], s0);
      s1 = wmma_bf16(aq[s], b1f[s], s1);
    }
#pragma unroll
    for (int r = 0; r < 8; ++r) {
      Sp[w][r + half * 8][ln]      = s0[r];
      Sp[w][r + half * 8][16 + ln] = s1[r];
    }
    __syncthreads();

    // ---- reduce 8 partials, scale + causal mask
    for (int e = tid; e < 512; e += 256) {
      const int row = e >> 5, key = e & 31;
      float acc = 0.f;
#pragma unroll
      for (int ww = 0; ww < 8; ++ww) acc += Sp[ww][row][key];
      acc *= scale;
      if (kb + key > qbase + row) acc = -3.0e38f;
      Sred[row][key] = acc;
    }
    __syncthreads();

    // ---- online softmax stats: wave 0, lane = (row, key-half)
    if (w == 0) {
      const int row = ln;
      const int kbase = half * 16;
      float m_old = mrow[row];
      float sv[16];
      float mloc = -3.0e38f;
#pragma unroll
      for (int k = 0; k < 16; ++k) {
        sv[k] = Sred[row][kbase + k];
        mloc = fmaxf(mloc, sv[k]);
      }
      float mpart = __shfl_xor(mloc, 16, 32);
      const float m_new = fmaxf(fmaxf(mloc, mpart), m_old);
      float lsum = 0.f;
#pragma unroll
      for (int k = 0; k < 16; ++k) {
        float p = __expf(sv[k] - m_new);
        lsum += p;
        P[row][kbase + k] = (__bf16)p;
      }
      lsum += __shfl_xor(lsum, 16, 32);
      if (half == 0) {
        float f = __expf(m_old - m_new);
        mrow[row] = m_new;
        lrow[row] = lrow[row] * f + lsum;
        frow[row] = f;
      }
    }
    __syncthreads();

    // ---- issue all V-fragment loads, rescale O, then WMMA accumulate
    v16bf bvf[8];
#pragma unroll
    for (int j = 0; j < 8; ++j)
      bvf[j] = *(const v16bf*)(vtp + (size_t)j * 16 * M_);

    float fr[8];
#pragma unroll
    for (int r = 0; r < 8; ++r) fr[r] = frow[r + half * 8];
    v16bf ap = *(const v16bf*)(&P[ln][koff]);
#pragma unroll
    for (int j = 0; j < 8; ++j) {
#pragma unroll
      for (int r = 0; r < 8; ++r) o[j][r] = o[j][r] * fr[r];
      o[j] = wmma_bf16(ap, bvf[j], o[j]);
    }
    __syncthreads();
  }

  // ---- epilogue: normalize by l and store f32 output
  float inv[8];
#pragma unroll
  for (int r = 0; r < 8; ++r) inv[r] = 1.0f / lrow[r + half * 8];
#pragma unroll
  for (int j = 0; j < 8; ++j) {
    const int dcol = dbase + j * 16 + ln;
    float* op = out + (rowoff + half * 8) * D_ + dcol;
#pragma unroll
    for (int r = 0; r < 8; ++r) op[(size_t)r * D_] = o[j][r] * inv[r];
  }
}

// ---------------------------------------------------------------------------
extern "C" void kernel_launch(void* const* d_in, const int* in_sizes, int n_in,
                              void* d_out, int out_size, void* d_ws, size_t ws_size,
                              hipStream_t stream) {
  (void)in_sizes; (void)n_in; (void)out_size; (void)ws_size;
  const float* x  = (const float*)d_in[0];
  const float* wq = (const float*)d_in[1];
  const float* wk = (const float*)d_in[2];
  const float* wv = (const float*)d_in[3];
  float* out = (float*)d_out;

  unsigned short* ws = (unsigned short*)d_ws;
  const size_t nX = (size_t)B_ * M_ * D_;   // 8,388,608
  const size_t nW = (size_t)D_ * D_;        // 1,048,576
  unsigned short* xb  = ws;
  unsigned short* wqb = xb + nX;
  unsigned short* wkb = wqb + nW;
  unsigned short* wvb = wkb + nW;
  unsigned short* Qr  = wvb + nW;
  unsigned short* Kr  = Qr + nX;
  unsigned short* VT  = Kr + nX;

  cvt_f32_bf16<<<1024, 256, 0, stream>>>(x,  xb,  (long)(nX / 4));
  cvt_f32_bf16<<<256,  256, 0, stream>>>(wq, wqb, (long)(nW / 4));
  cvt_f32_bf16<<<256,  256, 0, stream>>>(wk, wkb, (long)(nW / 4));
  cvt_f32_bf16<<<256,  256, 0, stream>>>(wv, wvb, (long)(nW / 4));

  qkv_rope<<<dim3(64, 64), 256, 0, stream>>>(xb, wqb, wkb, wvb, Qr, Kr, VT);
  attn_flash<<<dim3(M_ / 16, B_), 256, 0, stream>>>(Qr, Kr, VT, out);
}
